// NMF2D_12309376270915
// MI455X (gfx1250) — matrix operand order repaired and checked by reading, not compiled, not run on hardware
//
#include <hip/hip_runtime.h>
#include <hip/hip_bf16.h>

// ---------------- types ----------------
typedef __bf16 bf16;
typedef bf16  bf16x8  __attribute__((ext_vector_type(8)));
typedef bf16  bf16x16 __attribute__((ext_vector_type(16)));
typedef float f32x4   __attribute__((ext_vector_type(4)));
typedef float f32x8   __attribute__((ext_vector_type(8)));

#define NB   16     // batches
#define NC   512    // channels
#define NN   4096   // spatial (64*64)
#define NR   64     // rank
#define EPSV 1e-6f
#define TLDS 272    // LDS row stride (bf16 elems) for transpose staging: 544 B, 16B-aligned

// ---------------- WMMA helpers ----------------
__device__ __forceinline__ bf16x16 cat8(bf16x8 lo, bf16x8 hi) {
    return __builtin_shufflevector(lo, hi, 0,1,2,3,4,5,6,7,8,9,10,11,12,13,14,15);
}

// A fragment: 16x32 bf16 tile from row-major [M][ldk] (K contiguous).
__device__ __forceinline__ bf16x16 load_frag_a(const bf16* base, int ldk, int lane) {
    int row = lane & 15;
    int kb  = (lane < 16) ? 0 : 8;
    const bf16* p = base + (size_t)row * ldk + kb;
    bf16x8 lo = *(const bf16x8*)p;
    bf16x8 hi = *(const bf16x8*)(p + 16);
    return cat8(lo, hi);
}

// B fragment: 32x16 bf16 tile, sourced from B^T stored row-major [N][ldk] (K contiguous).
__device__ __forceinline__ bf16x16 load_frag_b(const bf16* baseT, int ldk, int lane) {
    int col = lane & 15;
    int kb  = (lane < 16) ? 0 : 8;
    const bf16* p = baseT + (size_t)col * ldk + kb;
    bf16x8 lo = *(const bf16x8*)p;
    bf16x8 hi = *(const bf16x8*)(p + 16);
    return cat8(lo, hi);
}

__device__ __forceinline__ f32x8 wmma_bf16(bf16x16 a, bf16x16 b, f32x8 c) {
    return __builtin_amdgcn_wmma_f32_16x16x32_bf16(false, a, false, b, (short)0, c, false, false);
}

__device__ __forceinline__ f32x8 zero8() {
    f32x8 z = {0.f,0.f,0.f,0.f,0.f,0.f,0.f,0.f};
    return z;
}

// ---------------- kernel 1: convert x -> bf16 (both orientations) ----------------
// grid (NN/32, NC/32, NB), block (32, 8)
__global__ __launch_bounds__(256) void k_xconv(const float* __restrict__ x,
                                               bf16* __restrict__ xbf,
                                               bf16* __restrict__ xT) {
    __shared__ float tile[32][33];
    int b = blockIdx.z, c0 = blockIdx.y * 32, n0 = blockIdx.x * 32;
    int tx = threadIdx.x, ty = threadIdx.y;
#pragma unroll
    for (int i = 0; i < 4; ++i) {
        int c = c0 + ty + i * 8;
        size_t idx = ((size_t)(b * NC + c)) * NN + n0 + tx;
        float v = x[idx];
        xbf[idx] = (bf16)v;
        tile[ty + i * 8][tx] = v;
    }
    __syncthreads();
#pragma unroll
    for (int i = 0; i < 4; ++i) {
        int n = n0 + ty + i * 8;
        xT[((size_t)(b * NN + n)) * NC + c0 + tx] = (bf16)tile[tx][ty + i * 8];
    }
}

// ---------------- kernel 2: normalize bases over channel dim ----------------
// grid (NB), block 256
__global__ __launch_bounds__(256) void k_norm_bases(const float* __restrict__ binit,
                                                    float* __restrict__ bases,
                                                    bf16* __restrict__ bases_bf,
                                                    bf16* __restrict__ basesT_bf) {
    __shared__ float red[256];
    __shared__ float scale[NR];
    int b = blockIdx.x, t = threadIdx.x;
    int r = t & 63, seg = t >> 6;
    const float* src = binit + (size_t)b * NC * NR;
    float s = 0.f;
    for (int c = seg; c < NC; c += 4) { float v = src[c * NR + r]; s += v * v; }
    red[t] = s;
    __syncthreads();
    if (t < NR) {
        float tot = red[t] + red[t + 64] + red[t + 128] + red[t + 192];
        scale[t] = 1.0f / fmaxf(sqrtf(tot), 1e-12f);
    }
    __syncthreads();
    for (int e = t; e < NC * NR; e += 256) {
        int c = e >> 6, rr = e & 63;
        float v = src[e] * scale[rr];
        bases[(size_t)b * NC * NR + e]               = v;
        bases_bf[(size_t)b * NC * NR + e]            = (bf16)v;
        basesT_bf[(size_t)b * NR * NC + rr * NC + c] = (bf16)v;
    }
}

// ---------------- kernel 3: num[b][n][r] = sum_c x[c][n]*bases[c][r] ----------------
// Oriented M=r (A from basesT in LDS), N=n (B from xT rows) so stores are contiguous in r.
// grid (NN/128, NB), block 256 (8 waves, each wave: 16 n-cols x all 64 r-rows)
__global__ __launch_bounds__(256) void k_gemm_num(const bf16* __restrict__ xT,
                                                  const bf16* __restrict__ basesT,
                                                  float* __restrict__ num) {
    __shared__ __align__(16) bf16 As[NR * 256];   // 32 KB: basesT K-chunk
    int b = blockIdx.y;
    int w = threadIdx.x >> 5, lane = threadIdx.x & 31;
    int n0 = blockIdx.x * 128 + w * 16;
    f32x8 acc[4] = { zero8(), zero8(), zero8(), zero8() };

    for (int kc = 0; kc < NC; kc += 256) {
        __syncthreads();
        const bf16* g = basesT + (size_t)b * NR * NC + kc;
        for (int j = threadIdx.x; j < 2048; j += 256) {     // 64 rows * 32 uint4 each
            int r = j >> 5, kk = (j & 31) * 8;
            *(uint4*)(As + r * 256 + kk) = *(const uint4*)(g + (size_t)r * NC + kk);
        }
        __syncthreads();
#pragma unroll
        for (int k0 = 0; k0 < 256; k0 += 32) {
            bf16x16 bb = load_frag_b(xT + ((size_t)(b * NN + n0)) * NC + kc + k0, NC, lane);
            bf16x16 af[4];
#pragma unroll
            for (int mt = 0; mt < 4; ++mt)
                af[mt] = load_frag_a(As + (mt * 16) * 256 + k0, 256, lane);
#pragma unroll
            for (int mt = 0; mt < 4; ++mt)
                acc[mt] = wmma_bf16(af[mt], bb, acc[mt]);
        }
    }
    int n = n0 + (lane & 15), mb = (lane < 16) ? 0 : 8;
#pragma unroll
    for (int mt = 0; mt < 4; ++mt) {
        float* p = num + ((size_t)(b * NN + n)) * NR + mt * 16 + mb;
        f32x4 lo, hi;
#pragma unroll
        for (int v = 0; v < 4; ++v) { lo[v] = acc[mt][v]; hi[v] = acc[mt][v + 4]; }
        *(f32x4*)(p)     = lo;
        *(f32x4*)(p + 4) = hi;
    }
}

// ---------------- kernel 4: softmax over r, write coef (fp32/bf16/T-bf16) ----------------
// grid (NB*NN/256), block 256
__global__ __launch_bounds__(256) void k_softmax(const float* __restrict__ num,
                                                 float* __restrict__ coef,
                                                 bf16* __restrict__ coef_bf,
                                                 bf16* __restrict__ coefT_bf) {
    int i = blockIdx.x * 256 + threadIdx.x;       // row index over b*n
    int b = i >> 12, n = i & (NN - 1);
    const float* src = num + (size_t)i * NR;
    float vals[NR];
    float mx = -3.4e38f;
#pragma unroll
    for (int r = 0; r < NR; ++r) { vals[r] = src[r]; mx = fmaxf(mx, vals[r]); }
    float s = 0.f;
#pragma unroll
    for (int r = 0; r < NR; ++r) { vals[r] = __expf(vals[r] - mx); s += vals[r]; }
    float inv = 1.0f / s;
#pragma unroll
    for (int r = 0; r < NR; ++r) {
        float v = vals[r] * inv;
        coef[(size_t)i * NR + r]    = v;
        coef_bf[(size_t)i * NR + r] = (bf16)v;
        coefT_bf[((size_t)(b * NR + r)) * NN + n] = (bf16)v;
    }
}

// ---------------- kernel 5: gram: out[b][64][64] = mT * mT^T, mT: [b][64][K] ----------------
// grid (NB), block 256 (8 waves x 2 tiles)
__global__ __launch_bounds__(256) void k_gram(const bf16* __restrict__ mT, int K,
                                              bf16* __restrict__ out_bf) {
    int b = blockIdx.x;
    int w = threadIdx.x >> 5, lane = threadIdx.x & 31;
    const bf16* base = mT + (size_t)b * NR * K;
#pragma unroll
    for (int tlo = 0; tlo < 2; ++tlo) {
        int T = w + tlo * 8;              // 0..15 -> 4x4 tile grid
        int mt = T >> 2, nt = T & 3;
        f32x8 acc = zero8();
        for (int k0 = 0; k0 < K; k0 += 32) {
            bf16x16 a  = load_frag_a(base + (size_t)(mt * 16) * K + k0, K, lane);
            bf16x16 bb = load_frag_b(base + (size_t)(nt * 16) * K + k0, K, lane);
            acc = wmma_bf16(a, bb, acc);
        }
        int col = nt * 16 + (lane & 15), mb = (lane < 16) ? 0 : 8;
#pragma unroll
        for (int v = 0; v < 8; ++v)
            out_bf[(size_t)b * NR * NR + (mt * 16 + mb + v) * NR + col] = (bf16)acc[v];
    }
}

// ---------------- kernel 6: fused multiplicative update ----------------
// den^T[r][row] = gram[r][q] * m[row][q];  m_new = m_old * numer / (den + eps)
// coef: rows=4096, tld=4096 ; bases: rows=512, tld=512
// grid (rows/256, NB), block 256. wave: mt = w&3 (r-tile), half = w>>2 (128 rows)
// Transposed bf16 shadow written via LDS staging -> coalesced uint4 rows.
__global__ __launch_bounds__(256) void k_update(const bf16* __restrict__ gram,
                                                const float* __restrict__ numer,
                                                float* __restrict__ m32,
                                                bf16* __restrict__ mbf,
                                                bf16* __restrict__ mTbf,
                                                int tld) {
    __shared__ __align__(16) bf16 Ls[NR * TLDS];  // 64 x 256 (stride 272) ~34 KB
    int rows = gridDim.x * 256;
    int b = blockIdx.y;
    int w = threadIdx.x >> 5, lane = threadIdx.x & 31;
    int mt = w & 3;
    int nblk = blockIdx.x * 256;
    int nloc0 = (w >> 2) * 128;
    int n0 = nblk + nloc0;
    f32x8 acc[8] = { zero8(), zero8(), zero8(), zero8(), zero8(), zero8(), zero8(), zero8() };

    const bf16* A = gram + (size_t)b * NR * NR + (size_t)(mt * 16) * NR;
#pragma unroll
    for (int k0 = 0; k0 < NR; k0 += 32) {
        bf16x16 a = load_frag_a(A + k0, NR, lane);
        bf16x16 bb[8];
#pragma unroll
        for (int t = 0; t < 8; ++t)
            bb[t] = load_frag_b(mbf + ((size_t)(b * rows + n0 + t * 16)) * NR + k0, NR, lane);
#pragma unroll
        for (int t = 0; t < 8; ++t)
            acc[t] = wmma_bf16(a, bb[t], acc[t]);
    }
    __syncthreads();   // all waves done reading mbf rows before any wave rewrites them

    int mb = (lane < 16) ? 0 : 8;
    int rb = mt * 16 + mb;
#pragma unroll
    for (int t = 0; t < 8; ++t) {
        int nloc = nloc0 + t * 16 + (lane & 15);
        int n = nblk + nloc;
        size_t base = ((size_t)(b * rows + n)) * NR + rb;
        f32x4 nlo = *(const f32x4*)(numer + base);
        f32x4 nhi = *(const f32x4*)(numer + base + 4);
        f32x4 clo = *(const f32x4*)(m32 + base);
        f32x4 chi = *(const f32x4*)(m32 + base + 4);
        f32x4 olo, ohi;
#pragma unroll
        for (int v = 0; v < 4; ++v) {
            olo[v] = clo[v] * nlo[v] / (acc[t][v]     + EPSV);
            ohi[v] = chi[v] * nhi[v] / (acc[t][v + 4] + EPSV);
        }
        *(f32x4*)(m32 + base)     = olo;
        *(f32x4*)(m32 + base + 4) = ohi;
        bf16x8 cb;
#pragma unroll
        for (int v = 0; v < 4; ++v) { cb[v] = (bf16)olo[v]; cb[v + 4] = (bf16)ohi[v]; }
        *(bf16x8*)(mbf + base) = cb;
#pragma unroll
        for (int v = 0; v < 4; ++v) {
            Ls[(rb + v) * TLDS + nloc]     = (bf16)olo[v];
            Ls[(rb + 4 + v) * TLDS + nloc] = (bf16)ohi[v];
        }
    }
    __syncthreads();
    // coalesced write-out of the 64 x 256 transposed tile
    for (int j = threadIdx.x; j < 2048; j += 256) {   // 64 rows * 32 uint4
        int r = j >> 5, q = (j & 31) * 8;
        *(uint4*)(mTbf + ((size_t)(b * NR + r)) * tld + nblk + q) =
            *(const uint4*)(Ls + r * TLDS + q);
    }
}

// ---------------- kernel 7: num_b partials: part[ks][b][c][r] = sum_{n in chunk} x[c][n]*coef[n][r] ----------------
// grid (NC/256, 8, NB), block 256. wave: mt = w&3 (r-tile), half=w>>2 (128 c-cols)
__global__ __launch_bounds__(256) void k_gemm_numb(const bf16* __restrict__ coefT,
                                                   const bf16* __restrict__ xbf,
                                                   float* __restrict__ part) {
    int b = blockIdx.z, ks = blockIdx.y;
    int w = threadIdx.x >> 5, lane = threadIdx.x & 31;
    int mt = w & 3;
    int c0 = blockIdx.x * 256 + (w >> 2) * 128;
    size_t kbase = (size_t)ks * (NN / 8);
    f32x8 acc[8] = { zero8(), zero8(), zero8(), zero8(), zero8(), zero8(), zero8(), zero8() };

    const bf16* A = coefT + ((size_t)(b * NR + mt * 16)) * NN + kbase;
#pragma unroll 2
    for (int k0 = 0; k0 < NN / 8; k0 += 32) {
        bf16x16 a = load_frag_a(A + k0, NN, lane);
        bf16x16 bb[8];
#pragma unroll
        for (int t = 0; t < 8; ++t)
            bb[t] = load_frag_b(xbf + ((size_t)(b * NC + c0 + t * 16)) * NN + kbase + k0, NN, lane);
#pragma unroll
        for (int t = 0; t < 8; ++t)
            acc[t] = wmma_bf16(a, bb[t], acc[t]);
    }
    int mb = (lane < 16) ? 0 : 8, col = lane & 15;
    float* dst = part + (size_t)ks * NB * NC * NR;
#pragma unroll
    for (int t = 0; t < 8; ++t) {
        int c = c0 + t * 16 + col;
        size_t base = ((size_t)(b * NC + c)) * NR + mt * 16 + mb;
        f32x4 lo, hi;
#pragma unroll
        for (int v = 0; v < 4; ++v) { lo[v] = acc[t][v]; hi[v] = acc[t][v + 4]; }
        *(f32x4*)(dst + base)     = lo;
        *(f32x4*)(dst + base + 4) = hi;
    }
}

// grid (NB*NC*NR/256), block 256: deterministic 8-way reduce
__global__ __launch_bounds__(256) void k_nbreduce(const float* __restrict__ part,
                                                  float* __restrict__ num_b) {
    size_t i = (size_t)blockIdx.x * 256 + threadIdx.x;
    const size_t stride = (size_t)NB * NC * NR;
    float s = 0.f;
#pragma unroll
    for (int ks = 0; ks < 8; ++ks) s += part[(size_t)ks * stride + i];
    num_b[i] = s;
}

// ---------------- kernel 8: out[b][c][n] = sum_r bases[c][r]*coef[n][r] ----------------
// out^T-oriented tiles: M=n, N=c -> contiguous stores along n. grid (NN/128, NC/128, NB)
__global__ __launch_bounds__(256) void k_out(const bf16* __restrict__ coef_bf,
                                             const bf16* __restrict__ bases_bf,
                                             float* __restrict__ out) {
    int b = blockIdx.z;
    int w = threadIdx.x >> 5, lane = threadIdx.x & 31;
    int n0 = blockIdx.x * 128 + w * 16;
    int c0 = blockIdx.y * 128;
    f32x8 acc[8] = { zero8(), zero8(), zero8(), zero8(), zero8(), zero8(), zero8(), zero8() };

    const bf16* A = coef_bf + ((size_t)(b * NN + n0)) * NR;
#pragma unroll
    for (int k0 = 0; k0 < NR; k0 += 32) {
        bf16x16 a = load_frag_a(A + k0, NR, lane);
        bf16x16 bb[8];
#pragma unroll
        for (int t = 0; t < 8; ++t)
            bb[t] = load_frag_b(bases_bf + ((size_t)(b * NC + c0 + t * 16)) * NR + k0, NR, lane);
#pragma unroll
        for (int t = 0; t < 8; ++t)
            acc[t] = wmma_bf16(a, bb[t], acc[t]);
    }
    int mb = (lane < 16) ? 0 : 8, col = lane & 15;
#pragma unroll
    for (int t = 0; t < 8; ++t) {
        int c = c0 + t * 16 + col;
        float* p = out + ((size_t)(b * NC + c)) * NN + n0 + mb;
        f32x4 lo, hi;
#pragma unroll
        for (int v = 0; v < 4; ++v) { lo[v] = acc[t][v]; hi[v] = acc[t][v + 4]; }
        *(f32x4*)(p)     = lo;
        *(f32x4*)(p + 4) = hi;
    }
}

// ---------------- host launcher ----------------
extern "C" void kernel_launch(void* const* d_in, const int* in_sizes, int n_in,
                              void* d_out, int out_size, void* d_ws, size_t ws_size,
                              hipStream_t stream) {
    (void)in_sizes; (void)n_in; (void)out_size;
    const float* x     = (const float*)d_in[0];   // [16,512,64,64]
    const float* binit = (const float*)d_in[1];   // [16,512,64]
    float* out = (float*)d_out;

    char* ws = (char*)d_ws;
    size_t off = 0;
    auto alloc = [&](size_t bytes) { char* p = ws + off; off += (bytes + 255) & ~(size_t)255; return p; };

    bf16*  xbf      = (bf16*)alloc((size_t)NB * NC * NN * 2);   // 64 MiB
    bf16*  xT       = (bf16*)alloc((size_t)NB * NN * NC * 2);   // 64 MiB
    float* coef32   = (float*)alloc((size_t)NB * NN * NR * 4);  // 16 MiB
    bf16*  coef_bf  = (bf16*)alloc((size_t)NB * NN * NR * 2);
    bf16*  coefT_bf = (bf16*)alloc((size_t)NB * NR * NN * 2);
    float* bases32  = (float*)alloc((size_t)NB * NC * NR * 4);
    bf16*  bases_bf = (bf16*)alloc((size_t)NB * NC * NR * 2);
    bf16*  basesT_bf= (bf16*)alloc((size_t)NB * NR * NC * 2);
    float* num      = (float*)alloc((size_t)NB * NN * NR * 4);  // 16 MiB
    float* num_b    = (float*)alloc((size_t)NB * NC * NR * 4);
    float* nbpart   = (float*)alloc((size_t)8 * NB * NC * NR * 4); // 16 MiB
    bf16*  BtB_bf   = (bf16*)alloc((size_t)NB * NR * NR * 2);
    bf16*  CtC_bf   = (bf16*)alloc((size_t)NB * NR * NR * 2);
    if (off > ws_size) return;   // workspace too small; nothing sane to do

    // --- prep ---
    k_xconv<<<dim3(NN / 32, NC / 32, NB), dim3(32, 8), 0, stream>>>(x, xbf, xT);
    k_norm_bases<<<NB, 256, 0, stream>>>(binit, bases32, bases_bf, basesT_bf);

    // --- init coef = softmax(x^T bases) ---
    k_gemm_num<<<dim3(NN / 128, NB), 256, 0, stream>>>(xT, basesT_bf, num);
    k_softmax<<<NB * NN / 256, 256, 0, stream>>>(num, coef32, coef_bf, coefT_bf);

    // --- 6 multiplicative-update steps ---
    for (int step = 0; step < 6; ++step) {
        if (step > 0)   // step 0 reuses num (bases unchanged since init gemm)
            k_gemm_num<<<dim3(NN / 128, NB), 256, 0, stream>>>(xT, basesT_bf, num);
        k_gram<<<NB, 256, 0, stream>>>(basesT_bf, NC, BtB_bf);
        k_update<<<dim3(NN / 256, NB), 256, 0, stream>>>(BtB_bf, num, coef32, coef_bf, coefT_bf, NN);

        k_gemm_numb<<<dim3(NC / 256, 8, NB), 256, 0, stream>>>(coefT_bf, xbf, nbpart);
        k_nbreduce<<<NB * NC * NR / 256, 256, 0, stream>>>(nbpart, num_b);
        k_gram<<<NB, 256, 0, stream>>>(coefT_bf, NN, CtC_bf);
        k_update<<<dim3(NC / 256, NB), 256, 0, stream>>>(CtC_bf, num_b, bases32, bases_bf, basesT_bf, NC);
    }

    // --- final coef refresh ---
    k_gemm_num<<<dim3(NN / 128, NB), 256, 0, stream>>>(xT, basesT_bf, num);
    k_gram<<<NB, 256, 0, stream>>>(basesT_bf, NC, BtB_bf);
    k_update<<<dim3(NN / 256, NB), 256, 0, stream>>>(BtB_bf, num, coef32, coef_bf, coefT_bf, NN);

    // --- out = bases @ coef^T ---
    k_out<<<dim3(NN / 128, NC / 128, NB), 256, 0, stream>>>(coef_bf, bases_bf, out);
}
